// LGP_22892175688205
// MI455X (gfx1250) — matrix operand order repaired and checked by reading, not compiled, not run on hardware
//
#include <hip/hip_runtime.h>

// ---------------------------------------------------------------------------
// Shapes (fixed by the reference): B=4, N=4096, C=384, K=16
// ---------------------------------------------------------------------------
#define KNN_K   16
#define C_DIM   384
#define C_P1    385      // C+1 (dist column)
#define C_PAD   416      // C+1 padded to multiple of 32 for bf16 WMMA K-dim

typedef __attribute__((ext_vector_type(2)))  float    v2f;
typedef __attribute__((ext_vector_type(8)))  float    v8f;
typedef __attribute__((ext_vector_type(16))) __bf16   v16bf;
typedef __attribute__((ext_vector_type(4)))  unsigned v4u;
typedef __attribute__((ext_vector_type(8)))  unsigned v8u;

__device__ __forceinline__ unsigned short f2bf(float f) {
    union { float f; unsigned u; } v; v.f = f;
    unsigned r = v.u + 0x7FFFu + ((v.u >> 16) & 1u);   // round-to-nearest-even
    return (unsigned short)(r >> 16);
}

// ---------------------------------------------------------------------------
// Kernel 1: KNN via V_WMMA_F32_16X16X4_F32 + Tensor Data Mover staging.
//
// One wave per 16 query rows; 8 waves/block all share one batch (256 row
// tiles per batch, divisible by 8), so a single TENSOR_LOAD_TO_LDS DMA pulls
// the whole batch xyz slab (4096*3 f32 = 48KB) into LDS once per block.
// Selection key = |c|^2 - 2 q.c  (|q|^2 is row-constant and sqrt monotone ->
// identical top-k; downstream mean over K is order-invariant, so no sort).
// Lanes 0-15 / 16-31 keep independent top-16 lists over candidate halves,
// merged through LDS at the end.
// ---------------------------------------------------------------------------
__device__ __forceinline__ void topk_insert(float d, int id,
                                            float* bd, int* bi,
                                            float& maxval, int& maxpos) {
    if (d < maxval) {
        float nm = -3.0e38f; int np = 0;
        #pragma unroll
        for (int i = 0; i < KNN_K; ++i) {
            if (i == maxpos) { bd[i] = d; bi[i] = id; }
            float v = bd[i];
            if (v > nm) { nm = v; np = i; }
        }
        maxval = nm; maxpos = np;
    }
}

__global__ void __launch_bounds__(256)
knn_kernel(const float* __restrict__ xyz, int* __restrict__ knn, int B, int N) {
    const int lane = threadIdx.x & 31;
    const int wave = threadIdx.x >> 5;
    const int rowTiles = N >> 4;                 // 256 tiles of 16 query rows
    const int wg = blockIdx.x * 8 + wave;
    const int b  = wg / rowTiles;
    const int rt = wg % rowTiles;

    __shared__ float sx[12288];                  // 48KB: whole batch xyz
    __shared__ float scratch[8][512];            // 16KB: per-wave tile / merge

    // ---- one TDM DMA per block: xyz[b] (N*3 f32) -> LDS ----
    if (threadIdx.x < 32) {
        const unsigned lds_off = (unsigned)(size_t)(void*)sx;   // low 32b = LDS offset
        const unsigned long long ga =
            (unsigned long long)(size_t)(const void*)(xyz + (size_t)b * N * 3);
        const unsigned ne = (unsigned)(N * 3);                  // 12288 elements

        v4u g0;
        g0[0] = 1u;                                             // count=1, user mode
        g0[1] = lds_off;                                        // lds_addr
        g0[2] = (unsigned)(ga & 0xFFFFFFFFu);                   // global_addr[31:0]
        g0[3] = (unsigned)((ga >> 32) & 0x1FFFFFFu) | (2u << 30); // addr[56:32], type=2

        v8u g1;
        g1[0] = 0x00020000u;              // workgroup_mask=0, data_size=2 (4B)
        g1[1] = (ne & 0xFFFFu) << 16;     // tensor_dim0[15:0]  (bits 63:48)
        g1[2] = (ne >> 16) | (1u << 16);  // tensor_dim0[31:16], tensor_dim1=1
        g1[3] = (ne & 0xFFFFu) << 16;     // tile_dim0 = ne     (bits 127:112)
        g1[4] = 0u;                       // tile_dim1=0, tile_dim2=0 (unused)
        g1[5] = ne;                       // tensor_dim0_stride[31:0]
        g1[6] = 0u;                       // stride hi / dim1_stride lo
        g1[7] = 0u;

        asm volatile("tensor_load_to_lds %0, %1" :: "s"(g0), "s"(g1) : "memory");
        __builtin_amdgcn_s_wait_tensorcnt(0);
    }
    __syncthreads();

    const int r16 = lane & 15;
    const int sub = lane >> 4;                   // 0 or 1 (lane half)
    const int row = (rt << 4) + r16;

    const float qx = sx[row * 3 + 0];
    const float qy = sx[row * 3 + 1];
    const float qz = sx[row * 3 + 2];

    // A fragment (16x4 f32): lane<16 holds K=0,1 ; lane>=16 holds K=2,3
    v2f a;
    a.x = (lane < 16) ? (-2.0f * qx) : (-2.0f * qz);
    a.y = (lane < 16) ? (-2.0f * qy) : 1.0f;

    float bd[KNN_K]; int bi[KNN_K];
    #pragma unroll
    for (int i = 0; i < KNN_K; ++i) { bd[i] = 3.0e38f; bi[i] = 0; }
    float maxval = 3.0e38f; int maxpos = 0;

    float* tileS = &scratch[wave][0];            // 256 f32 during the loop

    for (int t = 0; t < rowTiles; ++t) {
        const int col3 = (((t << 4) + r16)) * 3;
        const float cx = sx[col3 + 0];
        const float cy = sx[col3 + 1];
        const float cz = sx[col3 + 2];
        const float sq = cx * cx + cy * cy + cz * cz;

        // B fragment (4x16 f32): lane<16 holds K=0,1 ; lane>=16 holds K=2,3
        v2f bm;
        bm.x = (lane < 16) ? cx : cz;
        bm.y = (lane < 16) ? cy : sq;

        v8f acc = {0.f,0.f,0.f,0.f,0.f,0.f,0.f,0.f};
        acc = __builtin_amdgcn_wmma_f32_16x16x4_f32(
                  false, a, false, bm, (short)0, acc, false, false);

        // C layout: VGPR r -> M = r + 8*sub, N = r16
        #pragma unroll
        for (int r = 0; r < 8; ++r)
            tileS[(sub * 8 + r) * 16 + r16] = acc[r];
        __syncthreads();

        // lane scans its row (r16), its half of the 16 candidate columns
        #pragma unroll
        for (int j = 0; j < 8; ++j) {
            const int cc = sub * 8 + j;
            topk_insert(tileS[r16 * 16 + cc], (t << 4) + cc,
                        bd, bi, maxval, maxpos);
        }
        __syncthreads();
    }

    // merge upper-half lists into lower lanes (reuse scratch region)
    float* mrgD = &scratch[wave][0];
    int*   mrgI = (int*)&scratch[wave][256];
    if (lane >= 16) {
        #pragma unroll
        for (int i = 0; i < KNN_K; ++i) {
            mrgD[r16 * KNN_K + i] = bd[i];
            mrgI[r16 * KNN_K + i] = bi[i];
        }
    }
    __syncthreads();
    if (lane < 16) {
        #pragma unroll
        for (int i = 0; i < KNN_K; ++i)
            topk_insert(mrgD[r16 * KNN_K + i], mrgI[r16 * KNN_K + i],
                        bd, bi, maxval, maxpos);
        int* out = knn + ((size_t)b * N + row) * KNN_K;
        #pragma unroll
        for (int i = 0; i < KNN_K; ++i) out[i] = bi[i];
    }
}

// ---------------------------------------------------------------------------
// Kernel 2: gather + layernorm + Gaussian weight + mean over K.
// One block (16 waves) per point; wave k owns neighbor k; wave-shuffle
// reductions for the C=384 layernorm; ds_add_f32 LDS atomics for mean over K.
// Emits the fused (C+1) row zero-padded to 416 in bf16 for the MLP GEMMs.
// ---------------------------------------------------------------------------
__global__ void __launch_bounds__(512)
aggregate_kernel(const float* __restrict__ xyz, const float* __restrict__ feat,
                 const int* __restrict__ knn, unsigned short* __restrict__ Xpad,
                 int B, int N) {
    const int tid  = threadIdx.x;
    const int wave = tid >> 5;      // neighbor index k = 0..15
    const int lane = tid & 31;
    const int p = blockIdx.x;       // global point id (b*N + n)
    const int b = p / N;
    const int n = p % N;

    __shared__ int   idx_s[KNN_K];
    __shared__ float nx[KNN_K][3];
    __shared__ float mu[3], sg[3];
    __shared__ float fused[C_P1];

    if (tid < C_P1) fused[tid] = 0.0f;
    if (tid < KNN_K) idx_s[tid] = knn[(size_t)p * KNN_K + tid];
    __syncthreads();
    if (tid < KNN_K) {
        const float* xp = xyz + ((size_t)b * N + idx_s[tid]) * 3;
        nx[tid][0] = xp[0]; nx[tid][1] = xp[1]; nx[tid][2] = xp[2];
    }
    __syncthreads();
    if (tid < 3) {
        float s = 0.f;
        #pragma unroll
        for (int k = 0; k < KNN_K; ++k) s += nx[k][tid];
        const float m = s * (1.0f / 16.0f);
        float v = 0.f;
        #pragma unroll
        for (int k = 0; k < KNN_K; ++k) { float d = nx[k][tid] - m; v += d * d; }
        mu[tid] = m;
        sg[tid] = sqrtf(v * (1.0f / 15.0f)) + 1e-6f;   // ddof=1 + eps
    }
    __syncthreads();

    const int j = idx_s[wave];
    const float o0 = (nx[wave][0] - mu[0]) / sg[0];
    const float o1 = (nx[wave][1] - mu[1]) / sg[1];
    const float o2 = (nx[wave][2] - mu[2]) / sg[2];
    const float dist = sqrtf(o0 * o0 + o1 * o1 + o2 * o2);
    const float wgt  = expf(-0.5f * dist);

    const float* fc = feat + ((size_t)b * N + n) * C_DIM;
    const float* fn = feat + ((size_t)b * N + j) * C_DIM;

    float dl[C_DIM / 32];
    float s = 0.f;
    #pragma unroll
    for (int i = 0; i < C_DIM / 32; ++i) {
        const int c = lane + 32 * i;
        dl[i] = fn[c] - fc[c];
        s += dl[i];
    }
    #pragma unroll
    for (int m = 16; m >= 1; m >>= 1) s += __shfl_xor(s, m, 32);
    const float mean = s * (1.0f / (float)C_DIM);
    float v = 0.f;
    #pragma unroll
    for (int i = 0; i < C_DIM / 32; ++i) { float d = dl[i] - mean; v += d * d; }
    #pragma unroll
    for (int m = 16; m >= 1; m >>= 1) v += __shfl_xor(v, m, 32);
    const float rs = rsqrtf(v * (1.0f / (float)C_DIM) + 1e-5f);

    const float scale = wgt * rs * (1.0f / 16.0f);
    #pragma unroll
    for (int i = 0; i < C_DIM / 32; ++i)
        atomicAdd(&fused[lane + 32 * i], (dl[i] - mean) * scale);
    if (lane == 0) atomicAdd(&fused[C_DIM], dist * (1.0f / 16.0f));
    __syncthreads();

    if (tid < C_PAD) {
        const float vv = (tid < C_P1) ? fused[tid] : 0.0f;
        Xpad[(size_t)p * C_PAD + tid] = f2bf(vv);
    }
}

// ---------------------------------------------------------------------------
// Kernel 3: repack W1 (385x384) / W2 (384x384) transposed + padded to bf16 so
// WMMA B-fragments load contiguous K-pairs.
// ---------------------------------------------------------------------------
__global__ void pack_weights(const float* __restrict__ W1,
                             const float* __restrict__ W2,
                             unsigned short* __restrict__ W1t,
                             unsigned short* __restrict__ W2t) {
    const int t = blockIdx.x * blockDim.x + threadIdx.x;
    const int n1 = C_DIM * C_PAD;
    if (t < n1) {
        const int n = t / C_PAD, k = t % C_PAD;
        const float v = (k < C_P1) ? W1[(size_t)k * C_DIM + n] : 0.0f;
        W1t[(size_t)n * C_PAD + k] = f2bf(v);
    }
    const int t2 = t - n1;
    if (t2 >= 0 && t2 < C_DIM * C_DIM) {
        const int n = t2 / C_DIM, k = t2 % C_DIM;
        W2t[(size_t)n * C_DIM + k] = f2bf(W2[(size_t)k * C_DIM + n]);
    }
}

// ---------------------------------------------------------------------------
// bf16 fragment loaders per the CDNA5 16-bit A(16x32)/B(32x16) VGPR layouts.
// ---------------------------------------------------------------------------
__device__ __forceinline__ v16bf load_fragA(const unsigned short* __restrict__ base,
                                            int stride, int row, int k0, int lane) {
    union { unsigned u[8]; v16bf v; } f;
    const int khalf = (lane >> 4) * 8;
    const unsigned short* rp = base + (size_t)row * stride;
    #pragma unroll
    for (int vi = 0; vi < 8; ++vi) {
        const int k = k0 + ((vi < 4) ? 0 : 16) + khalf + ((vi & 3) * 2);
        f.u[vi] = *(const unsigned int*)(rp + k);
    }
    return f.v;
}

__device__ __forceinline__ v16bf load_fragB(const unsigned short* __restrict__ base,
                                            int stride, int col, int k0, int lane) {
    union { unsigned u[8]; v16bf v; } f;
    const int koff = (lane >> 4) * 8;
    const unsigned short* cp = base + (size_t)col * stride + k0 + koff;
    #pragma unroll
    for (int vi = 0; vi < 8; ++vi)
        f.u[vi] = *(const unsigned int*)(cp + ((vi < 4) ? 0 : 16) + (vi & 3) * 2);
    return f.v;
}

// ---------------------------------------------------------------------------
// Kernel 4: bf16 WMMA GEMM (+bias, optional exact GELU). One wave per 16x16
// output tile; K loop of 16x16x32 bf16 WMMAs with f32 accumulation.
// Writes bf16 (for H, consumed by GEMM2) or f32 (final output).
// ---------------------------------------------------------------------------
__global__ void __launch_bounds__(256)
gemm_bf16_kernel(const unsigned short* __restrict__ A,
                 const unsigned short* __restrict__ Bt,
                 const float* __restrict__ bias,
                 unsigned short* __restrict__ outBf,
                 float* __restrict__ outF32,
                 int Nc, int Kp, int gelu) {
    const int lane = threadIdx.x & 31;
    const int wave = threadIdx.x >> 5;
    const int colTiles = Nc >> 4;
    const int wg = blockIdx.x * 8 + wave;
    const int tM = wg / colTiles;
    const int tN = wg % colTiles;
    const int row = tM * 16 + (lane & 15);
    const int col = tN * 16 + (lane & 15);

    v8f acc = {0.f,0.f,0.f,0.f,0.f,0.f,0.f,0.f};
    for (int k0 = 0; k0 < Kp; k0 += 32) {
        v16bf fa = load_fragA(A,  Kp, row, k0, lane);
        v16bf fb = load_fragB(Bt, Kp, col, k0, lane);
        acc = __builtin_amdgcn_wmma_f32_16x16x32_bf16(
                  false, fa, false, fb, (short)0, acc, false, false);
    }

    const float bc = bias[col];
    const int mbase = tM * 16 + ((lane >> 4) * 8);   // C layout: M = r + 8*(lane/16)
    #pragma unroll
    for (int r = 0; r < 8; ++r) {
        float x = acc[r] + bc;
        if (gelu) x = 0.5f * x * (1.0f + erff(x * 0.70710678118654752f));
        const int rr = mbase + r;
        if (outBf) outBf[(size_t)rr * Nc + col] = f2bf(x);
        else       outF32[(size_t)rr * Nc + col] = x;
    }
}

// ---------------------------------------------------------------------------
// Launch
// ---------------------------------------------------------------------------
extern "C" void kernel_launch(void* const* d_in, const int* in_sizes, int n_in,
                              void* d_out, int out_size, void* d_ws, size_t ws_size,
                              hipStream_t stream) {
    const float* xyz  = (const float*)d_in[0];
    const float* feat = (const float*)d_in[1];
    const float* W1   = (const float*)d_in[2];
    const float* b1   = (const float*)d_in[3];
    const float* W2   = (const float*)d_in[4];
    const float* b2   = (const float*)d_in[5];

    const int B  = 4;
    const int BN = in_sizes[0] / 3;     // B*N
    const int N  = BN / B;              // 4096
    const int M  = BN;                  // GEMM rows

    // workspace partition (256B aligned)
    char* ws = (char*)d_ws;
    size_t off = 0;
    auto take = [&](size_t bytes) {
        void* p = ws + off;
        off = (off + bytes + 255) & ~(size_t)255;
        return p;
    };
    int*            knn  = (int*)           take((size_t)BN * KNN_K * sizeof(int));
    unsigned short* Xpad = (unsigned short*)take((size_t)M * C_PAD * 2);
    unsigned short* W1t  = (unsigned short*)take((size_t)C_DIM * C_PAD * 2);
    unsigned short* W2t  = (unsigned short*)take((size_t)C_DIM * C_DIM * 2);
    unsigned short* H    = (unsigned short*)take((size_t)M * C_DIM * 2);
    (void)ws_size;

    // 1) KNN: B*(N/16) waves, 8 waves/block (divides evenly: 1024/8; all 8
    //    waves of a block share one batch since 256 % 8 == 0)
    {
        const int waves = B * (N >> 4);
        knn_kernel<<<waves / 8, 256, 0, stream>>>(xyz, knn, B, N);
    }
    // 2) weight repack (bf16, transposed, padded)
    {
        const int total = C_DIM * C_PAD + C_DIM * C_DIM;
        pack_weights<<<(total + 255) / 256, 256, 0, stream>>>(W1, W2, W1t, W2t);
    }
    // 3) gather + LN + weight + mean over K -> Xpad (bf16, M x 416)
    aggregate_kernel<<<BN, 512, 0, stream>>>(xyz, feat, knn, Xpad, B, N);

    // 4) H = gelu(Xpad @ W1 + b1)   (M x 384, bf16)
    {
        const int waves = (M >> 4) * (C_DIM >> 4);   // 1024*24, divisible by 8
        gemm_bf16_kernel<<<waves / 8, 256, 0, stream>>>(
            Xpad, W1t, b1, H, nullptr, C_DIM, C_PAD, 1);
    }
    // 5) out = H @ W2 + b2          (M x 384, f32 -> d_out)
    {
        const int waves = (M >> 4) * (C_DIM >> 4);
        gemm_bf16_kernel<<<waves / 8, 256, 0, stream>>>(
            H, W2t, b2, nullptr, (float*)d_out, C_DIM, C_DIM, 0);
    }
}
